// ScaledDotProduct_33646773796895
// MI455X (gfx1250) — matrix operand order repaired and checked by reading, not compiled, not run on hardware
//
#include <hip/hip_runtime.h>
#include <math.h>

typedef __attribute__((ext_vector_type(16))) _Float16 v16h;
typedef __attribute__((ext_vector_type(8)))  _Float16 v8h;
typedef __attribute__((ext_vector_type(2)))  _Float16 v2h;
typedef __attribute__((ext_vector_type(8)))  float    v8f;

#define S_LEN 2048
#define DK    64
#define BK    64             // keys processed per iteration
#define NKT   (S_LEN / BK)   // 32 key tiles
#define WAVES 4              // waves per workgroup
#define QTILE (WAVES * 16)   // q rows per workgroup
#define KT_STRIDE 72         // halves per K-tile row (144B, keeps 16B align)
#define VT_STRIDE 72         // halves per transposed-V row (144B, keeps 16B align)
#define TILE_ELTS (BK * DK)  // 4096 f32 per staged tile

__device__ __forceinline__ v8h cvt8(float4 a, float4 b) {
    v8h r;
    r[0] = (_Float16)a.x; r[1] = (_Float16)a.y; r[2] = (_Float16)a.z; r[3] = (_Float16)a.w;
    r[4] = (_Float16)b.x; r[5] = (_Float16)b.y; r[6] = (_Float16)b.z; r[7] = (_Float16)b.w;
    return r;
}

__device__ __forceinline__ v16h cat8(v8h lo, v8h hi) {
    v16h r;
#pragma unroll
    for (int e = 0; e < 8; ++e) { r[e] = lo[e]; r[e + 8] = hi[e]; }
    return r;
}

// VALU lane permutation within rows of 16 (v_permlane16_b32), nibble selectors.
__device__ __forceinline__ float permlane16_f(float x, unsigned s0, unsigned s1) {
    unsigned xi = __builtin_bit_cast(unsigned, x);
    unsigned r  = __builtin_amdgcn_permlane16(xi, xi, s0, s1, false, false);
    return __builtin_bit_cast(float, r);
}

// max across the 16-lane row (xor 1,2,4,8 butterflies), no DS traffic
__device__ __forceinline__ float rowmax16(float x) {
    x = fmaxf(x, permlane16_f(x, 0x67452301u, 0xEFCDAB89u)); // xor 1
    x = fmaxf(x, permlane16_f(x, 0x45670123u, 0xCDEF89ABu)); // xor 2
    x = fmaxf(x, permlane16_f(x, 0x32107654u, 0xBA98FEDCu)); // xor 4
    x = fmaxf(x, permlane16_f(x, 0xFEDCBA98u, 0x76543210u)); // xor 8
    return x;
}

// CDNA5 async global->LDS copy, tracked by ASYNCcnt.
__device__ __forceinline__ void async_ld_b128(unsigned lds_off, const void* g) {
    asm volatile("global_load_async_to_lds_b128 %0, %1, off"
                 :: "v"(lds_off), "v"((unsigned long long)(size_t)g)
                 : "memory");
}
__device__ __forceinline__ unsigned lds_off_of(const void* p) {
    return (unsigned)(size_t)p;   // flat LDS aperture: low 32 bits = DS offset
}

__global__ __launch_bounds__(WAVES * 32)
void sdpa_fa_f16_kernel(const float* __restrict__ Q,
                        const float* __restrict__ K,
                        const float* __restrict__ V,
                        const int*   __restrict__ mask,
                        float* __restrict__ O,
                        int h_heads)
{
    __shared__ float    Kst[TILE_ELTS];          // raw f32 K tile (async dest)
    __shared__ float    Vst[TILE_ELTS];          // raw f32 V tile (async dest)
    __shared__ _Float16 Kt[BK * KT_STRIDE];      // K tile, row-major f16
    __shared__ _Float16 Vt[DK * VT_STRIDE];      // V tile transposed: Vt[d][k]
    __shared__ _Float16 Plds[WAVES * 16 * BK];   // per-wave probability tiles

    const int tid  = threadIdx.x;
    const int lane = tid & 31;
    const int wave = tid >> 5;
    const int half = lane >> 4;   // 0: lanes 0-15, 1: lanes 16-31
    const int l16  = lane & 15;

    const int bh = blockIdx.y;
    const int b  = bh / h_heads;
    const int q0 = blockIdx.x * QTILE + wave * 16;
    const int qrow = q0 + l16;

    const size_t baseKV = (size_t)bh * S_LEN * DK;
    const size_t baseQ  = baseKV + (size_t)qrow * DK;

    // ---- Q A-fragments, fp32 -> fp16 once, 1/sqrt(dk) folded in ----
    v16h qa[2];
#pragma unroll
    for (int f = 0; f < 2; ++f) {
        const int dbase = f * 32;
        const float4* c0 = reinterpret_cast<const float4*>(Q + baseQ + dbase + half * 8);
        const float4* c1 = reinterpret_cast<const float4*>(Q + baseQ + dbase + 16 + half * 8);
        float4 a0 = c0[0], a1 = c0[1];
        float4 b0 = c1[0], b1 = c1[1];
        v16h q;
        q[0]  = (_Float16)(a0.x * 0.125f); q[1]  = (_Float16)(a0.y * 0.125f);
        q[2]  = (_Float16)(a0.z * 0.125f); q[3]  = (_Float16)(a0.w * 0.125f);
        q[4]  = (_Float16)(a1.x * 0.125f); q[5]  = (_Float16)(a1.y * 0.125f);
        q[6]  = (_Float16)(a1.z * 0.125f); q[7]  = (_Float16)(a1.w * 0.125f);
        q[8]  = (_Float16)(b0.x * 0.125f); q[9]  = (_Float16)(b0.y * 0.125f);
        q[10] = (_Float16)(b0.z * 0.125f); q[11] = (_Float16)(b0.w * 0.125f);
        q[12] = (_Float16)(b1.x * 0.125f); q[13] = (_Float16)(b1.y * 0.125f);
        q[14] = (_Float16)(b1.z * 0.125f); q[15] = (_Float16)(b1.w * 0.125f);
        qa[f] = q;
    }

    // all-ones B fragment: row sums of P via WMMA
    v16h ones;
#pragma unroll
    for (int e = 0; e < 16; ++e) ones[e] = (_Float16)1.0f;

    float m_run[8], l_run[8];
    v8f acc[4];
#pragma unroll
    for (int v = 0; v < 8; ++v) { m_run[v] = -INFINITY; l_run[v] = 0.0f; }
#pragma unroll
    for (int n = 0; n < 4; ++n)
#pragma unroll
        for (int v = 0; v < 8; ++v) acc[n][v] = 0.0f;

    _Float16* Pw = Plds + wave * (16 * BK);

    // copy slice (async): row kk, 32 contiguous f32 at cb
    const int kk  = tid >> 1;
    const int cb  = (tid & 1) * 32;
    const int slice = kk * DK + cb;
    // V transpose-staging slice: k-pair vk, 16 d's at vd
    const int vk = (tid & 31) * 2;
    const int vd = (tid >> 5) * 16;

    const unsigned koff = lds_off_of(Kst + slice);
    const unsigned voff = lds_off_of(Vst + slice);

    // ---- prologue: async-issue tile 0 (16 x b128 per lane) ----
    {
        const float* kg = K + baseKV + slice;
        const float* vg = V + baseKV + slice;
#pragma unroll
        for (int i = 0; i < 8; ++i) {
            async_ld_b128(koff + i * 16, kg + i * 4);
            async_ld_b128(voff + i * 16, vg + i * 4);
        }
    }

    for (int kt = 0; kt < NKT; ++kt) {
        const int k0 = kt * BK;

        asm volatile("s_wait_asynccnt 0x0" ::: "memory");  // this wave's tile landed
        __syncthreads();   // all waves landed; previous f16 tiles fully consumed

        // ---- stage f32 LDS -> f16 Kt (row-major) ----
        {
            const float4* kp = reinterpret_cast<const float4*>(Kst + slice);
            v8h* krow = reinterpret_cast<v8h*>(Kt + kk * KT_STRIDE + cb);
            krow[0] = cvt8(kp[0], kp[1]);
            krow[1] = cvt8(kp[2], kp[3]);
            krow[2] = cvt8(kp[4], kp[5]);
            krow[3] = cvt8(kp[6], kp[7]);
        }
        // ---- stage f32 LDS -> f16 Vt (transposed, packed k-pairs) ----
        {
            const float* r0 = Vst + vk * DK + vd;
            const float* r1 = r0 + DK;
            const float4* a = reinterpret_cast<const float4*>(r0);
            const float4* c = reinterpret_cast<const float4*>(r1);
            float x0[16] = { a[0].x, a[0].y, a[0].z, a[0].w, a[1].x, a[1].y, a[1].z, a[1].w,
                             a[2].x, a[2].y, a[2].z, a[2].w, a[3].x, a[3].y, a[3].z, a[3].w };
            float x1[16] = { c[0].x, c[0].y, c[0].z, c[0].w, c[1].x, c[1].y, c[1].z, c[1].w,
                             c[2].x, c[2].y, c[2].z, c[2].w, c[3].x, c[3].y, c[3].z, c[3].w };
#pragma unroll
            for (int e = 0; e < 16; ++e) {
                v2h pk; pk[0] = (_Float16)x0[e]; pk[1] = (_Float16)x1[e];
                *reinterpret_cast<v2h*>(Vt + (vd + e) * VT_STRIDE + vk) = pk;
            }
        }
        __syncthreads();

        // ---- async-issue the NEXT tile; overlaps the whole compute phase ----
        {
            const int ktn = (kt + 1) & (NKT - 1);     // wraps harmlessly at end
            const float* kg = K + baseKV + (size_t)ktn * TILE_ELTS + slice;
            const float* vg = V + baseKV + (size_t)ktn * TILE_ELTS + slice;
#pragma unroll
            for (int i = 0; i < 8; ++i) {
                async_ld_b128(koff + i * 16, kg + i * 4);
                async_ld_b128(voff + i * 16, vg + i * 4);
            }
        }

        // ---- scores S = (Q/8) K^T for 4 key sub-tiles ----
        v8f s[4];
#pragma unroll
        for (int n = 0; n < 4; ++n) {
            const int j = k0 + n * 16 + l16;
            const _Float16* kr = Kt + (n * 16 + l16) * KT_STRIDE + half * 8;
            v16h kb_lo = cat8(*reinterpret_cast<const v8h*>(kr),
                              *reinterpret_cast<const v8h*>(kr + 16));        // d in [0,32)
            v16h kb_hi = cat8(*reinterpret_cast<const v8h*>(kr + 32),
                              *reinterpret_cast<const v8h*>(kr + 48));        // d in [32,64)
            v8f c = {0.f, 0.f, 0.f, 0.f, 0.f, 0.f, 0.f, 0.f};
            c = __builtin_amdgcn_wmma_f32_16x16x32_f16(false, qa[0], false, kb_lo, (short)0, c, false, false);
            c = __builtin_amdgcn_wmma_f32_16x16x32_f16(false, qa[1], false, kb_hi, (short)0, c, false, false);
            const float madd = (mask[b * S_LEN + j] == 0) ? -1e-10f : 0.0f;
#pragma unroll
            for (int v = 0; v < 8; ++v) s[n][v] = c[v] + madd;
        }

        // ---- online softmax: one rowmax butterfly per 64 keys ----
        float p[4][8], alpha[8];
#pragma unroll
        for (int v = 0; v < 8; ++v) {
            float t = fmaxf(fmaxf(s[0][v], s[1][v]), fmaxf(s[2][v], s[3][v]));
            t = rowmax16(t);
            const float mnew = fmaxf(m_run[v], t);
#pragma unroll
            for (int n = 0; n < 4; ++n) p[n][v] = __expf(s[n][v] - mnew);
            alpha[v] = __expf(m_run[v] - mnew);
            m_run[v] = mnew;
        }

        // ---- transpose P (D layout -> A layout) through per-wave LDS ----
#pragma unroll
        for (int v = 0; v < 8; ++v) {
            const int m = v + half * 8;
#pragma unroll
            for (int n = 0; n < 4; ++n)
                Pw[m * BK + n * 16 + l16] = (_Float16)p[n][v];
        }
        const _Float16* pr = Pw + l16 * BK;
        v16h pa0 = cat8(*reinterpret_cast<const v8h*>(pr + half * 8),
                        *reinterpret_cast<const v8h*>(pr + 16 + half * 8));   // keys [0,32)
        v16h pa1 = cat8(*reinterpret_cast<const v8h*>(pr + 32 + half * 8),
                        *reinterpret_cast<const v8h*>(pr + 48 + half * 8));   // keys [32,64)

        // ---- row sums of P with WMMAs against the all-ones matrix ----
        v8f zc = {0.f, 0.f, 0.f, 0.f, 0.f, 0.f, 0.f, 0.f};
        v8f ts = __builtin_amdgcn_wmma_f32_16x16x32_f16(false, pa0, false, ones, (short)0, zc, false, false);
        ts     = __builtin_amdgcn_wmma_f32_16x16x32_f16(false, pa1, false, ones, (short)0, ts, false, false);
#pragma unroll
        for (int v = 0; v < 8; ++v) {
            l_run[v] = l_run[v] * alpha[v] + ts[v];
#pragma unroll
            for (int n2 = 0; n2 < 4; ++n2) acc[n2][v] *= alpha[v];
        }

        // ---- O += P V via 4 N-tiles x 2 K-chunks of transposed-V fragments ----
#pragma unroll
        for (int n2 = 0; n2 < 4; ++n2) {
            const _Float16* vr = Vt + (n2 * 16 + l16) * VT_STRIDE;
            v16h vb_lo = cat8(*reinterpret_cast<const v8h*>(vr + half * 8),
                              *reinterpret_cast<const v8h*>(vr + 16 + half * 8));  // k in [0,32)
            v16h vb_hi = cat8(*reinterpret_cast<const v8h*>(vr + 32 + half * 8),
                              *reinterpret_cast<const v8h*>(vr + 48 + half * 8)); // k in [32,64)
            acc[n2] = __builtin_amdgcn_wmma_f32_16x16x32_f16(false, pa0, false, vb_lo, (short)0, acc[n2], false, false);
            acc[n2] = __builtin_amdgcn_wmma_f32_16x16x32_f16(false, pa1, false, vb_hi, (short)0, acc[n2], false, false);
        }
    }

    // ---- epilogue: normalize by l and store ----
#pragma unroll
    for (int v = 0; v < 8; ++v) {
        const float inv = 1.0f / l_run[v];
        const int m = v + half * 8;
        const size_t orow = ((size_t)bh * S_LEN + (size_t)(q0 + m)) * DK;
#pragma unroll
        for (int n2 = 0; n2 < 4; ++n2)
            O[orow + n2 * 16 + l16] = acc[n2][v] * inv;
    }
}

extern "C" void kernel_launch(void* const* d_in, const int* in_sizes, int n_in,
                              void* d_out, int out_size, void* d_ws, size_t ws_size,
                              hipStream_t stream) {
    const float* Q    = (const float*)d_in[0];
    const float* K    = (const float*)d_in[1];
    const float* V    = (const float*)d_in[2];
    const int*   mask = (const int*)d_in[3];
    float* O = (float*)d_out;

    const int BH = in_sizes[0] / (S_LEN * DK);   // bs*h = 64
    const int bs = in_sizes[3] / S_LEN;          // 4
    const int h  = BH / bs;                      // 16

    dim3 grid(S_LEN / QTILE, BH);
    dim3 block(WAVES * 32);
    sdpa_fa_f16_kernel<<<grid, block, 0, stream>>>(Q, K, V, mask, O, h);
}
